// Attention_6545530159487
// MI455X (gfx1250) — matrix operand order repaired and checked by reading, not compiled, not run on hardware
//
#include <hip/hip_runtime.h>
#include <hip/hip_bf16.h>

// ---------------------------------------------------------------------------
// MHA forward: B=4, N=1024, DIM=1024, H=16, DH=64 on gfx1250 (wave32).
// - all matmuls on v_wmma_f32_16x16x32_f16, operands fed by b128 vector loads
// - weights pre-transposed to f16 column-major; V stored d-major per head
// - shared A panels staged in LDS via GLOBAL_LOAD_ASYNC_TO_LDS (double buffer)
// - attn tile written LDS->HBM via GLOBAL_STORE_ASYNC_FROM_LDS, overlapped
//   with the P@V WMMA phase (S_ENDPGM implicitly waits all counters)
// ---------------------------------------------------------------------------

typedef __attribute__((ext_vector_type(16))) _Float16 v16h;
typedef __attribute__((ext_vector_type(8)))  _Float16 h8;
typedef __attribute__((ext_vector_type(4)))  _Float16 h4;
typedef __attribute__((ext_vector_type(8)))  float    v8f;

#define B_    4
#define N_    1024
#define DIM_  1024
#define H_    16
#define DH_   64
#define BN_   (B_ * N_)        // 4096 rows
#define SCALE_ 0.125f          // DH^-0.5

// ---- async global<->LDS support (guarded; sync fallback otherwise) --------
#if defined(__has_builtin)
#if __has_builtin(__builtin_amdgcn_global_load_async_to_lds_b64) &&            \
    __has_builtin(__builtin_amdgcn_global_store_async_from_lds_b128) &&        \
    __has_builtin(__builtin_amdgcn_s_wait_asynccnt)
#define USE_ASYNC 1
#endif
#endif
#ifndef USE_ASYNC
#define USE_ASYNC 0
#endif

#if USE_ASYNC
// gcc-style vectors to match the builtin prototypes exactly
typedef int vi2_ __attribute__((vector_size(2 * sizeof(int))));
typedef int vi4_ __attribute__((vector_size(4 * sizeof(int))));
typedef __attribute__((address_space(1))) vi2_ g_vi2;   // global int2
typedef __attribute__((address_space(3))) vi2_ l_vi2;   // LDS int2
typedef __attribute__((address_space(1))) vi4_ g_vi4;   // global int4
typedef __attribute__((address_space(3))) vi4_ l_vi4;   // LDS int4
#endif

// copy 8 bytes global -> LDS (async when available)
static __device__ inline void copy_g2l_b64(const _Float16* g, _Float16* l) {
#if USE_ASYNC
  __builtin_amdgcn_global_load_async_to_lds_b64((g_vi2*)g, (l_vi2*)l, 0, 0);
#else
  *(h4*)l = *(const h4*)g;
#endif
}

// copy 16 bytes LDS -> global (async when available)
static __device__ inline void copy_l2g_b128(float* g, const float* l) {
#if USE_ASYNC
  __builtin_amdgcn_global_store_async_from_lds_b128((g_vi4*)g, (l_vi4*)l, 0, 0);
#else
  *(float4*)g = *(const float4*)l;
#endif
}

static __device__ inline void wait_async_all() {
#if USE_ASYNC
  __builtin_amdgcn_s_wait_asynccnt(0);
#endif
}

// ---- WMMA fragment helpers (k order: contiguous k = 16h + s in A and B) -----------------
static __device__ inline v16h cat8(h8 lo, h8 hi) {
  return __builtin_shufflevector(lo, hi, 0, 1, 2, 3, 4, 5, 6, 7,
                                 8, 9, 10, 11, 12, 13, 14, 15);
}

// A fragment (16x32 f16) from row-major f16: two 16B loads (8 halves each).
static __device__ inline v16h frag_a_f16(const _Float16* __restrict__ A,
                                         int lda, int row, int kb) {
  const _Float16* p = A + (size_t)row * lda + kb;
  h8 lo = *(const h8*)p;
  h8 hi = *(const h8*)(p + 8);
  return cat8(lo, hi);
}

// B fragment (32x16 f16) from column-major f16 (Bt[n][k]): one 32B load.
static __device__ inline v16h frag_b_cm(const _Float16* __restrict__ Bt,
                                        int ldk, int col, int kg) {
  return *(const v16h*)(Bt + (size_t)col * ldk + kg);
}

static __device__ inline float swz_xor1(float x) {
  return __int_as_float(__builtin_amdgcn_ds_swizzle(__float_as_int(x), 0x041F));
}
static __device__ inline float swz_xor2(float x) {
  return __int_as_float(__builtin_amdgcn_ds_swizzle(__float_as_int(x), 0x081F));
}
static __device__ inline float swz_xor4(float x) {
  return __int_as_float(__builtin_amdgcn_ds_swizzle(__float_as_int(x), 0x101F));
}

// ---------------------------------------------------------------------------
// Prep A: x (f32, [4096x1024]) -> f16 same layout.  4 elems/thread.
// ---------------------------------------------------------------------------
__global__ __launch_bounds__(256) void cvt_x_kernel(
    const float* __restrict__ x, _Float16* __restrict__ x16) {
  const size_t i = ((size_t)blockIdx.x * 256 + threadIdx.x) * 4;
  float4 v = *(const float4*)(x + i);
  h4 o;
  o[0] = (_Float16)v.x; o[1] = (_Float16)v.y;
  o[2] = (_Float16)v.z; o[3] = (_Float16)v.w;
  *(h4*)(x16 + i) = o;
}

// ---------------------------------------------------------------------------
// Prep B: transpose + convert 1024x1024 f32 weights to f16 column-major.
// ---------------------------------------------------------------------------
__global__ __launch_bounds__(256) void transpose_w_kernel(
    const float* __restrict__ W0, const float* __restrict__ W1,
    const float* __restrict__ W2, const float* __restrict__ W3,
    _Float16* __restrict__ T0, _Float16* __restrict__ T1,
    _Float16* __restrict__ T2, _Float16* __restrict__ T3) {
  __shared__ float tile[32][33];
  const float* W = (blockIdx.z == 0) ? W0 : (blockIdx.z == 1) ? W1
                   : (blockIdx.z == 2) ? W2 : W3;
  _Float16* T = (blockIdx.z == 0) ? T0 : (blockIdx.z == 1) ? T1
                : (blockIdx.z == 2) ? T2 : T3;
  const int tx = threadIdx.x, ty = threadIdx.y;
  const int k0 = blockIdx.y * 32, n0 = blockIdx.x * 32;
#pragma unroll
  for (int i = 0; i < 4; ++i)
    tile[ty + i * 8][tx] = W[(size_t)(k0 + ty + i * 8) * DIM_ + n0 + tx];
  __syncthreads();
#pragma unroll
  for (int i = 0; i < 4; ++i)
    T[(size_t)(n0 + ty + i * 8) * DIM_ + k0 + tx] =
        (_Float16)tile[tx][ty + i * 8];
}

// ---------------------------------------------------------------------------
// Kernel 1: Q/K/V projection.  grid = (BN/16, 1024/256, 3), block = 128.
// The 16x32 A panel (shared by all 4 waves) is staged in LDS with async
// copies, double-buffered so the k+32 copy overlaps WMMA at k.
// ---------------------------------------------------------------------------
__global__ __launch_bounds__(128) void qkv_proj_kernel(
    const _Float16* __restrict__ x16,
    const _Float16* __restrict__ Wqt, const _Float16* __restrict__ Wkt,
    const _Float16* __restrict__ Wvt,
    _Float16* __restrict__ Qw, _Float16* __restrict__ Kw,
    _Float16* __restrict__ Vt) {
  __shared__ _Float16 apanel[2][16 * 32];   // 2 KB double buffer

  const int tid  = threadIdx.x;
  const int lane = tid & 31;
  const int wave = tid >> 5;
  const int rowL = lane & 15;
  const int hi   = lane >> 4;
  const int row0 = blockIdx.x * 16;
  const int n0   = blockIdx.y * 256 + wave * 64;
  const int z    = blockIdx.z;

  const _Float16* Wt = (z == 0) ? Wqt : (z == 1) ? Wkt : Wvt;

  // panel stage: 128 threads x 4 halves = 16x32 tile
  const int sidx = tid * 4;
  const int srow = sidx >> 5, scol = sidx & 31;
  const _Float16* gsrc = x16 + (size_t)(row0 + srow) * DIM_ + scol;

  copy_g2l_b64(gsrc, &apanel[0][sidx]);   // k0 = 0

  v8f acc[4] = {};
  for (int k0 = 0; k0 < DIM_; k0 += 32) {
    const int buf = (k0 >> 5) & 1;
    wait_async_all();
    __syncthreads();
    if (k0 + 32 < DIM_)
      copy_g2l_b64(gsrc + k0 + 32, &apanel[buf ^ 1][sidx]);

    v16h a = frag_a_f16(apanel[buf], 32, rowL, hi * 16);
    v16h bfr[4];
#pragma unroll
    for (int t = 0; t < 4; ++t) {
      const int col = n0 + t * 16 + rowL;
      bfr[t] = frag_b_cm(Wt, DIM_, col, k0 + hi * 16);
      __builtin_prefetch(Wt + (size_t)col * DIM_ + k0 + 64, 0, 3);
    }
#pragma unroll
    for (int t = 0; t < 4; ++t)
      acc[t] = __builtin_amdgcn_wmma_f32_16x16x32_f16(
          false, a, false, bfr[t], (short)0, acc[t], false, false);
  }

#pragma unroll
  for (int t = 0; t < 4; ++t) {
#pragma unroll
    for (int r = 0; r < 8; ++r) {
      const int orow = row0 + r + hi * 8;      // row in [0, BN)
      const int ocol = n0 + t * 16 + rowL;     // col in [0, H*DH)
      const int bb = orow >> 10, nn = orow & (N_ - 1);
      const int hh = ocol >> 6,  dd = ocol & (DH_ - 1);
      const size_t bh = (size_t)(bb * H_ + hh);
      const _Float16 val = (_Float16)acc[t][r];
      if (z == 2)
        Vt[(bh * DH_ + dd) * N_ + nn] = val;          // [B*H, DH, N]
      else if (z == 1)
        Kw[(bh * N_ + nn) * DH_ + dd] = val;          // [B*H, N, DH]
      else
        Qw[(bh * N_ + nn) * DH_ + dd] = val;
    }
  }
}

// ---------------------------------------------------------------------------
// Kernel 2: attention.  grid = (N/16, H, B), block = 128 (4 waves).
// Q fragments hoisted (register reuse over all 64 key tiles); 64KB LDS score
// tile; ds_swizzle softmax; attn written via async LDS->global stores that
// overlap the P@V WMMA phase; P@V B-fragments contiguous from d-major V.
// ---------------------------------------------------------------------------
__global__ __launch_bounds__(128) void attn_kernel(
    const _Float16* __restrict__ Qw, const _Float16* __restrict__ Kw,
    const _Float16* __restrict__ Vt,
    float* __restrict__ attn_out, _Float16* __restrict__ Ow) {
  __shared__ float S[16 * N_];   // 64 KB

  const int lane = threadIdx.x & 31;
  const int wave = threadIdx.x >> 5;
  const int rowL = lane & 15;
  const int hi   = lane >> 4;
  const int h = blockIdx.y, b = blockIdx.z;
  const int bh = b * H_ + h;
  const int q0 = blockIdx.x * 16;

  const _Float16* Qh = Qw + (size_t)bh * N_ * DH_;
  const _Float16* Kh = Kw + (size_t)bh * N_ * DH_;
  const _Float16* Vh = Vt + (size_t)bh * DH_ * N_;   // [DH, N]

  // ---- Phase 1: S = (Q Kt) * scale;  Q fragments loaded once ----
  const v16h aq0 = frag_a_f16(Qh, DH_, q0 + rowL, 0 + hi * 16);
  const v16h aq1 = frag_a_f16(Qh, DH_, q0 + rowL, 32 + hi * 16);
  for (int jt = wave; jt < N_ / 16; jt += 4) {
    const _Float16* kp = Kh + (size_t)(jt * 16 + rowL) * DH_ + hi * 16;
    v16h b0 = *(const v16h*)(kp);        // d = 0..31 slice
    v16h b1 = *(const v16h*)(kp + 32);   // d = 32..63 slice
    v8f acc = {};
    acc = __builtin_amdgcn_wmma_f32_16x16x32_f16(
        false, aq0, false, b0, (short)0, acc, false, false);
    acc = __builtin_amdgcn_wmma_f32_16x16x32_f16(
        false, aq1, false, b1, (short)0, acc, false, false);
#pragma unroll
    for (int r = 0; r < 8; ++r)
      S[(r + hi * 8) * N_ + jt * 16 + rowL] = acc[r] * SCALE_;
  }
  __syncthreads();

  // ---- Phase 2: row softmax (8 lanes per row, xor-swizzle reduction) ----
  {
    const int srow = wave * 4 + (lane >> 3);
    const int l8 = lane & 7;
    float* Sr = S + srow * N_;
    float m = -3.402823466e38f;
    for (int c = l8; c < N_; c += 8) m = fmaxf(m, Sr[c]);
    m = fmaxf(m, swz_xor1(m));
    m = fmaxf(m, swz_xor2(m));
    m = fmaxf(m, swz_xor4(m));
    float sum = 0.f;
    for (int c = l8; c < N_; c += 8) {
      float e = __expf(Sr[c] - m);
      Sr[c] = e;
      sum += e;
    }
    sum += swz_xor1(sum);
    sum += swz_xor2(sum);
    sum += swz_xor4(sum);
    const float inv = 1.0f / sum;
    for (int c = l8; c < N_; c += 8) Sr[c] *= inv;
  }
  __syncthreads();

  // ---- Phase 3: attn tile out via async LDS->global (overlaps phase 4) ----
  {
    float* gbase = attn_out + ((size_t)bh * N_ + q0) * N_;
    for (int i = threadIdx.x; i < 16 * N_ / 4; i += 128)
      copy_l2g_b128(gbase + i * 4, S + i * 4);
    // no wait: stores only read S; S_ENDPGM drains ASYNCcnt.
  }

  // ---- Phase 4: O = P @ V.  Wave w owns output cols [16w, 16w+16) of DH ----
  {
    const int dcol = wave * 16 + rowL;
    v8f acc = {};
    for (int k0 = 0; k0 < N_; k0 += 32) {
      // A from LDS: four ds_load_b128, cvt to f16
      const float* sp = S + rowL * N_ + k0 + hi * 16;
      float4 f0 = ((const float4*)sp)[0];
      float4 f1 = ((const float4*)sp)[1];
      float4 f2 = ((const float4*)(sp + 8))[0];
      float4 f3 = ((const float4*)(sp + 8))[1];
      v16h a;
      a[0] = (_Float16)f0.x;  a[1] = (_Float16)f0.y;
      a[2] = (_Float16)f0.z;  a[3] = (_Float16)f0.w;
      a[4] = (_Float16)f1.x;  a[5] = (_Float16)f1.y;
      a[6] = (_Float16)f1.z;  a[7] = (_Float16)f1.w;
      a[8] = (_Float16)f2.x;  a[9] = (_Float16)f2.y;
      a[10] = (_Float16)f2.z; a[11] = (_Float16)f2.w;
      a[12] = (_Float16)f3.x; a[13] = (_Float16)f3.y;
      a[14] = (_Float16)f3.z; a[15] = (_Float16)f3.w;
      // B[k=j][n=d] = Vh[d][j]: contiguous 16 halves from d-major V
      v16h bfr = *(const v16h*)(Vh + (size_t)dcol * N_ + k0 + hi * 16);
      acc = __builtin_amdgcn_wmma_f32_16x16x32_f16(
          false, a, false, bfr, (short)0, acc, false, false);
    }
#pragma unroll
    for (int r = 0; r < 8; ++r) {
      const int orow = q0 + r + hi * 8;
      Ow[(((size_t)b * N_ + orow) * H_ + h) * DH_ + wave * 16 + rowL] =
          (_Float16)acc[r];
    }
  }
}

// ---------------------------------------------------------------------------
// Kernel 3: out = Oheads @ Wo + bo.  grid = (BN/16, 1024/256), block = 128.
// Same async-staged A panel as kernel 1.
// ---------------------------------------------------------------------------
__global__ __launch_bounds__(128) void out_proj_kernel(
    const _Float16* __restrict__ Oin, const _Float16* __restrict__ Wot,
    const float* __restrict__ bo, float* __restrict__ out) {
  __shared__ _Float16 apanel[2][16 * 32];

  const int tid  = threadIdx.x;
  const int lane = tid & 31;
  const int wave = tid >> 5;
  const int rowL = lane & 15;
  const int hi   = lane >> 4;
  const int row0 = blockIdx.x * 16;
  const int n0   = blockIdx.y * 256 + wave * 64;

  const int sidx = tid * 4;
  const int srow = sidx >> 5, scol = sidx & 31;
  const _Float16* gsrc = Oin + (size_t)(row0 + srow) * (H_ * DH_) + scol;

  copy_g2l_b64(gsrc, &apanel[0][sidx]);

  v8f acc[4] = {};
  for (int k0 = 0; k0 < H_ * DH_; k0 += 32) {
    const int buf = (k0 >> 5) & 1;
    wait_async_all();
    __syncthreads();
    if (k0 + 32 < H_ * DH_)
      copy_g2l_b64(gsrc + k0 + 32, &apanel[buf ^ 1][sidx]);

    v16h a = frag_a_f16(apanel[buf], 32, rowL, hi * 16);
    v16h bfr[4];
#pragma unroll
    for (int t = 0; t < 4; ++t) {
      const int col = n0 + t * 16 + rowL;
      bfr[t] = frag_b_cm(Wot, DIM_, col, k0 + hi * 16);
      __builtin_prefetch(Wot + (size_t)col * DIM_ + k0 + 64, 0, 3);
    }
#pragma unroll
    for (int t = 0; t < 4; ++t)
      acc[t] = __builtin_amdgcn_wmma_f32_16x16x32_f16(
          false, a, false, bfr[t], (short)0, acc[t], false, false);
  }

#pragma unroll
  for (int t = 0; t < 4; ++t) {
    const int ocol = n0 + t * 16 + rowL;
    const float bias = bo[ocol];
#pragma unroll
    for (int r = 0; r < 8; ++r) {
      const int orow = row0 + r + hi * 8;
      out[(size_t)orow * DIM_ + ocol] = acc[t][r] + bias;
    }
  }
}

// ---------------------------------------------------------------------------
extern "C" void kernel_launch(void* const* d_in, const int* in_sizes, int n_in,
                              void* d_out, int out_size, void* d_ws,
                              size_t ws_size, hipStream_t stream) {
  const float* x  = (const float*)d_in[0];
  const float* Wq = (const float*)d_in[1];
  const float* Wk = (const float*)d_in[2];
  const float* Wv = (const float*)d_in[3];
  const float* Wo = (const float*)d_in[4];
  const float* bo = (const float*)d_in[5];

  float* out  = (float*)d_out;                       // [B, N, DIM]
  float* attn = out + (size_t)B_ * N_ * DIM_;        // [B, H, N, N]

  const size_t xe   = (size_t)BN_ * DIM_;            // 4M halves
  const size_t we   = (size_t)DIM_ * DIM_;           // 1M halves each
  const size_t qkve = (size_t)B_ * H_ * N_ * DH_;    // 4M halves each
  _Float16* x16 = (_Float16*)d_ws;
  _Float16* Wqt = x16 + xe;
  _Float16* Wkt = Wqt + we;
  _Float16* Wvt = Wkt + we;
  _Float16* Wot = Wvt + we;
  _Float16* Qw  = Wot + we;
  _Float16* Kw  = Qw + qkve;
  _Float16* Vt  = Kw + qkve;
  _Float16* Ow  = Vt + qkve;

  cvt_x_kernel<<<dim3((unsigned)(xe / (256 * 4))), 256, 0, stream>>>(x, x16);
  transpose_w_kernel<<<dim3(32, 32, 4), dim3(32, 8), 0, stream>>>(
      Wq, Wk, Wv, Wo, Wqt, Wkt, Wvt, Wot);
  qkv_proj_kernel<<<dim3(BN_ / 16, (H_ * DH_) / 256, 3), 128, 0, stream>>>(
      x16, Wqt, Wkt, Wvt, Qw, Kw, Vt);
  attn_kernel<<<dim3(N_ / 16, H_, B_), 128, 0, stream>>>(Qw, Kw, Vt, attn, Ow);
  out_proj_kernel<<<dim3(BN_ / 16, DIM_ / 256, 1), 128, 0, stream>>>(
      Ow, Wot, bo, out);
}
